// CIDERLoss_80315888435296
// MI455X (gfx1250) — matrix operand order repaired and checked by reading, not compiled, not run on hardware
//
#include <hip/hip_runtime.h>

// CDNA5 / gfx1250: wave32, WMMA 16x16x32 bf16, LDS-staged B operand, pipelined epilogue.
typedef __attribute__((ext_vector_type(16))) __bf16 v16bf;
typedef __attribute__((ext_vector_type(8)))  float  v8f;

#define NCLS   1000
#define CPAD   1024
#define DDIM   128
#define BROWS  65536
#define TINV   10.0f                    // 1/T, T = 0.1
#define LOG2E  1.4426950408889634f
#define LN2    0.6931471805599453f
#define SCALE2 (TINV * LOG2E)           // logits -> base-2 domain in one scale
#define NEGBIG (-1.0e30f)

#define WAVES_PB 16
#define THREADS  (WAVES_PB * 32)

// ws layout: float accum[2] at offset 0 (comp_sum, dis_sum); bf16 mu[1024][128] at +256B.

// Single-exp online logsumexp update (base-2). No fmin/fmax -> no canonicalize ops;
// neg+abs fold into v_exp_f32 source modifiers.
__device__ __forceinline__ void lse_update(float& m, float& s, float u)
{
    float t = u - m;
    float e = __builtin_amdgcn_exp2f(-__builtin_fabsf(t));  // exp2(min-max) <= 1
    bool gt = t > 0.0f;
    s = gt ? __builtin_fmaf(s, e, 1.0f) : (s + e);          // other term is exp2(0)=1
    m = gt ? u : m;
}

// Merge two (max, sum) pairs (base-2), one exp.
__device__ __forceinline__ void lse_combine(float& m, float& s, float mo, float so)
{
    float t = mo - m;
    float e = __builtin_amdgcn_exp2f(-__builtin_fabsf(t));
    bool gt = t > 0.0f;
    s = gt ? __builtin_fmaf(s, e, so) : __builtin_fmaf(so, e, s);
    m = gt ? mo : m;
}

// One 16x16 logits tile: 4 chained WMMAs over K=128.
__device__ __forceinline__ v8f wmma_tile(const __bf16* __restrict__ bp, const v16bf (&a)[4])
{
    v8f acc = {};
#pragma unroll
    for (int kk = 0; kk < 4; ++kk) {
        v16bf b = *(const v16bf*)(bp + kk * 32);             // ds_load_b128 x2
        acc = __builtin_amdgcn_wmma_f32_16x16x32_bf16(
                  false, a[kk], false, b, (short)0, acc, false, false);
    }
    return acc;
}

__global__ void cider_prep(const float* __restrict__ mu, __bf16* __restrict__ mub,
                           float* __restrict__ accum)
{
    int idx = blockIdx.x * blockDim.x + threadIdx.x;
    if (idx < 2) accum[idx] = 0.0f;
    if (idx < CPAD * DDIM) {
        int row = idx >> 7;                       // DDIM = 128
        __bf16 v = (__bf16)0.0f;
        if (row < NCLS) v = (__bf16)mu[idx];      // zero-pad rows 1000..1023
        mub[idx] = v;
    }
}

// Compactness: one wave = one 16-row z tile; 16 waves/block; 256 blocks.
__global__ __launch_bounds__(THREADS)
void cider_comp(const float* __restrict__ z, const int* __restrict__ tgt,
                const __bf16* __restrict__ mub, float* __restrict__ accum)
{
    __shared__ __bf16 smu[CPAD * DDIM];           // 256 KB of the 320 KB WGP LDS

    // Cooperative global -> LDS staging of the bf16 prototype matrix (16B chunks).
    {
        const uint4* src = (const uint4*)mub;
        uint4*       dst = (uint4*)smu;
        const int total = (CPAD * DDIM) / 8;      // 16384 uint4
#pragma unroll 4
        for (int i = threadIdx.x; i < total; i += THREADS)
            dst[i] = src[i];
    }
    __syncthreads();

    const int lane = threadIdx.x & 31;
    const int wave = threadIdx.x >> 5;
    const int N    = lane & 15;
    const int half = lane >> 4;
    const int base = (blockIdx.x * WAVES_PB + wave) * 16;

    // A fragments: 16x128 z tile, pre-scaled by (1/T)*log2e, bf16.
    // A layout: element s holds K = half*16+s (contiguous, matching the B loads).
    v16bf a[4];
    {
        const float* zr = z + (size_t)(base + N) * DDIM;
#pragma unroll
        for (int kk = 0; kk < 4; ++kk) {
            const float* p0 = zr + kk * 32 + half * 16;
#pragma unroll
            for (int i = 0; i < 8; ++i) {
                a[kk][i]     = (__bf16)(p0[i]      * SCALE2);
                a[kk][8 + i] = (__bf16)(p0[8 + i] * SCALE2);
            }
        }
    }

    float m[8], s[8];
#pragma unroll
    for (int r = 0; r < 8; ++r) { m[r] = NEGBIG; s[r] = 0.0f; }

    const __bf16* bbase = smu + (size_t)N * DDIM + half * 16;   // + c*16*DDIM per tile

    // Software-pipelined: compute tile c while consuming tile c-1 (hides the
    // WMMA->VALU RAW hazard and ds_load latency). Tiles 0..61 need no masking.
    v8f acc0 = wmma_tile(bbase, a);                              // tile 0
#pragma unroll 1
    for (int c = 1; c < 63; c += 2) {                            // 31 iterations
        v8f acc1 = wmma_tile(bbase + (size_t)c * (16 * DDIM), a);
#pragma unroll
        for (int r = 0; r < 8; ++r) lse_update(m[r], s[r], acc0[r]);
        acc0 = wmma_tile(bbase + (size_t)(c + 1) * (16 * DDIM), a);
#pragma unroll
        for (int r = 0; r < 8; ++r) lse_update(m[r], s[r], acc1[r]);
    }
    // acc0 now holds tile 62 (columns 992..1007): branchless mask of col >= 1000.
    {
        const bool colok = (992 + N) < NCLS;
#pragma unroll
        for (int r = 0; r < 8; ++r) lse_update(m[r], s[r], colok ? acc0[r] : NEGBIG);
    }

    // Row logsumexp: 16-lane butterfly (stays inside each half of the wave).
    float lsesum = 0.0f;                           // base-2 units, 8 rows per half
#pragma unroll
    for (int r = 0; r < 8; ++r) {
        float mm = m[r], ss = s[r];
#pragma unroll
        for (int off = 8; off >= 1; off >>= 1) {
            float mo = __shfl_xor(mm, off, 32);
            float so = __shfl_xor(ss, off, 32);
            lse_combine(mm, ss, mo, so);
        }
        lsesum += mm + __builtin_amdgcn_logf(ss);  // log2
    }
    lsesum += __shfl_xor(lsesum, 16, 32);          // all 16 rows of this wave

    // pos = z[row] . mu[target[row]] / T, fp32 cooperative dot (4 dims/lane).
    float possum = 0.0f;
#pragma unroll 2
    for (int i = 0; i < 16; ++i) {
        const int trow = tgt[base + i];            // wave-uniform -> scalar load
        const float*  zr = z   + (size_t)(base + i) * DDIM + lane * 4;
        const __bf16* mr = smu + (size_t)trow      * DDIM + lane * 4;
        float part;
        part = zr[0] * (float)mr[0];
        part = __builtin_fmaf(zr[1], (float)mr[1], part);
        part = __builtin_fmaf(zr[2], (float)mr[2], part);
        part = __builtin_fmaf(zr[3], (float)mr[3], part);
#pragma unroll
        for (int off = 16; off >= 1; off >>= 1) part += __shfl_xor(part, off, 32);
        possum += part;
    }
    possum *= TINV;

    if (lane == 0) atomicAdd(&accum[0], possum - LN2 * lsesum);   // Σ(pos - lse)
}

// Dispersion: mu @ mu.T / T, off-diagonal row logsumexp. 8 blocks x 8 waves.
__global__ __launch_bounds__(256)
void cider_dis(const __bf16* __restrict__ mub, float* __restrict__ accum)
{
    const int lane = threadIdx.x & 31;
    const int wave = threadIdx.x >> 5;
    const int N    = lane & 15;
    const int half = lane >> 4;
    const int base = (blockIdx.x * 8 + wave) * 16;

    v16bf a[4];
    {
        const __bf16* ar = mub + (size_t)(base + N) * DDIM;
#pragma unroll
        for (int kk = 0; kk < 4; ++kk) {
            const __bf16* p0 = ar + kk * 32 + half * 16;
#pragma unroll
            for (int i = 0; i < 8; ++i) {
                a[kk][i]     = p0[i];
                a[kk][8 + i] = p0[8 + i];
            }
        }
    }

    float m[8], s[8];
#pragma unroll
    for (int r = 0; r < 8; ++r) { m[r] = NEGBIG; s[r] = 0.0f; }

#pragma unroll 1
    for (int c = 0; c < 63; ++c) {
        v8f acc = wmma_tile(mub + ((size_t)c * 16 + N) * DDIM + half * 16, a);
        const int col = c * 16 + N;
        const bool colok = col < NCLS;
#pragma unroll
        for (int r = 0; r < 8; ++r) {
            const int row = base + half * 8 + r;
            float u = acc[r] * SCALE2;
            if (!colok || col == row) u = NEGBIG;   // branchless mask (pad cols + diagonal)
            lse_update(m[r], s[r], u);
        }
    }

    float lsesum = 0.0f;
#pragma unroll
    for (int r = 0; r < 8; ++r) {
        float mm = m[r], ss = s[r];
#pragma unroll
        for (int off = 8; off >= 1; off >>= 1) {
            float mo = __shfl_xor(mm, off, 32);
            float so = __shfl_xor(ss, off, 32);
            lse_combine(mm, ss, mo, so);
        }
        const int row = base + half * 8 + r;
        if (row < NCLS) lsesum += mm + __builtin_amdgcn_logf(ss);
    }
    lsesum += __shfl_xor(lsesum, 16, 32);
    if (lane == 0) atomicAdd(&accum[1], LN2 * lsesum);
}

__global__ void cider_fin(const float* __restrict__ accum, float* __restrict__ out)
{
    if (threadIdx.x == 0 && blockIdx.x == 0) {
        float loss_comp = -accum[0] / (float)BROWS;
        float loss_dis  =  accum[1] / (float)NCLS + __logf(1.0f / (float)(NCLS - 1));
        out[0] = 1.0f * loss_dis + 2.0f * loss_comp;   // ALPHA=1, LAMDA=2
    }
}

extern "C" void kernel_launch(void* const* d_in, const int* in_sizes, int n_in,
                              void* d_out, int out_size, void* d_ws, size_t ws_size,
                              hipStream_t stream)
{
    const float* z   = (const float*)d_in[0];   // [65536,128] f32
    const int*   tgt = (const int*)d_in[1];     // [65536]
    const float* mu  = (const float*)d_in[2];   // [1000,128] f32
    float* out = (float*)d_out;

    float*  accum = (float*)d_ws;
    __bf16* mub   = (__bf16*)((char*)d_ws + 256);   // 1024*128 bf16 = 256 KB

    cider_prep<<<(CPAD * DDIM) / 256, 256, 0, stream>>>(mu, mub, accum);
    cider_comp<<<BROWS / (16 * WAVES_PB), THREADS, 0, stream>>>(z, tgt, mub, accum);
    cider_dis<<<8, 256, 0, stream>>>(mub, accum);
    cider_fin<<<1, 1, 0, stream>>>(accum, out);
}
